// BLSTM_47339129536587
// MI455X (gfx1250) — matrix operand order repaired and checked by reading, not compiled, not run on hardware
//
#include <hip/hip_runtime.h>

typedef __attribute__((ext_vector_type(16))) _Float16 v16h;
typedef __attribute__((ext_vector_type(8)))  float    v8f;
typedef __attribute__((ext_vector_type(4)))  float    v4f;
typedef __attribute__((ext_vector_type(4)))  int      v4i;

static constexpr int kB  = 1024;  // batch
static constexpr int kT  = 512;   // time steps
static constexpr int kV  = 128;   // vocab
static constexpr int kH  = 128;   // embed dim
static constexpr int kHH = 64;    // half hidden
static constexpr int kNG = 256;   // 4*kHH gate columns

__device__ __forceinline__ float frcp_(float x) {
    return __builtin_amdgcn_rcpf(x);          // v_rcp_f32, no IEEE div expansion
}
__device__ __forceinline__ float sigmoidf_(float x) {
    return frcp_(1.0f + __expf(-x));
}
__device__ __forceinline__ float tanhf_(float x) {
    // 1 - 2/(e^{2x}+1): monotone, saturates to +-1 without inf/inf NaNs
    return 1.0f - 2.0f * frcp_(__expf(2.0f * x) + 1.0f);
}

// A-fragment indexing this code uses for 16-bit WMMA A (16x32), contiguous k order:
// element e of lane L holds M = L&15, K = kb + e + 16*(L>>4).
// We store h so each lane's 16 halves are contiguous: phys = frag*512 + L*16 + e.
__device__ __forceinline__ int h_phys(int m, int k) {
    int frag = k >> 5;
    int khw  = (k >> 4) & 1;
    int e    = k & 15;
    return frag * 512 + (m + 16 * khw) * 16 + e;
}

// ---------------------------------------------------------------------------
// Interleaved table: tbl[v][k][g] = sum_h emb[v][h] * W_ih[g*64+k][h]
// (folds embedding @ W_ih.T; [k][4] interleave -> one b128 per row/lane)
// grid (128, 2) x 256 threads: y=0 -> forward table, y=1 -> backward table
// ---------------------------------------------------------------------------
__global__ void blstm_prep_table(const float* __restrict__ emb,
                                 const float* __restrict__ Wih_f,
                                 const float* __restrict__ Wih_b,
                                 float* __restrict__ table_f,
                                 float* __restrict__ table_b) {
    const float* Wih = blockIdx.y ? Wih_b : Wih_f;
    float* tbl       = blockIdx.y ? table_b : table_f;
    int idx = blockIdx.x * blockDim.x + threadIdx.x;  // 0 .. 128*256-1
    int v   = idx >> 8;
    int col = idx & (kNG - 1);
    int k   = col >> 2;          // hidden index 0..63
    int g   = col & 3;           // gate 0..3 (i,f,g,o)
    const float* er = emb + v * kH;
    const float* wr = Wih + (g * kHH + k) * kH;
    float s = 0.f;
#pragma unroll 8
    for (int h = 0; h < kH; ++h) s = fmaf(er[h], wr[h], s);
    tbl[idx] = s;
}

__global__ void blstm_f32_to_f16(const float* __restrict__ in,
                                 _Float16* __restrict__ out, int n) {
    int i = blockIdx.x * blockDim.x + threadIdx.x;
    if (i < n) out[i] = (_Float16)in[i];
}

// ---------------------------------------------------------------------------
// LSTM scan. grid (B/16, 2 directions) x 128 threads (4 waves).
// Wave w owns hidden slice [16w,16w+16): computes i/f/g/o gate tiles for that
// slice (8 WMMAs from C=0), adds the token-table input term during the
// elementwise phase, keeps cell state c in registers; only h (f16, 2KB,
// double-buffered, A-fragment physical layout) round-trips LDS ->
// one barrier per step. Time loop unrolled x2 with fixed ping/pong buffers;
// direction handled by a strided token pointer (no per-step selects).
// ---------------------------------------------------------------------------
__global__ void __launch_bounds__(128, 1)
blstm_scan(const int* __restrict__ x,
           const float* __restrict__ table_f,
           const float* __restrict__ table_b,
           const _Float16* __restrict__ Whh_f,
           const _Float16* __restrict__ Whh_b,
           _Float16* __restrict__ h_final) {
    extern __shared__ char smem[];
    float*    s_table = (float*)(smem);                 // [128][64][4] 131072 B
    _Float16* s_W     = (_Float16*)(smem + 131072);     // [256][64]     32768 B
    int*      s_tok   = (int*)(smem + 163840);          // [512][16]     32768 B (time-major)
    _Float16* s_h0    = (_Float16*)(smem + 196608);     // phys layout    2048 B
    _Float16* s_h1    = (_Float16*)(smem + 198656);     // phys layout    2048 B

    const int tid = threadIdx.x;
    const int dir = blockIdx.y;
    const int b0  = blockIdx.x * 16;

    const float*    gtab = dir ? table_b : table_f;
    const _Float16* gW   = dir ? Whh_b : Whh_f;

    for (int i = tid; i < kV * kNG; i += 128) s_table[i] = gtab[i];
    for (int i = tid; i < kNG * kHH; i += 128) s_W[i] = gW[i];
    for (int i = tid; i < 16 * kT; i += 128) {          // coalesced read, LDS scatter
        int m = i >> 9, tt = i & (kT - 1);
        s_tok[tt * 16 + m] = x[(b0 + m) * kT + tt];
    }
    for (int i = tid; i < 16 * kHH; i += 128) {
        s_h0[i] = (_Float16)0.f;
        s_h1[i] = (_Float16)0.f;
    }
    __syncthreads();

    const int lane = tid & 31;
    const int wave = tid >> 5;            // 0..3, owns hidden slice 16*wave..+16
    const int ln16 = lane & 15;
    const int kh   = lane >> 4;           // K-half selector per WMMA layout
    const int col  = wave * 16 + ln16;    // hidden index this lane owns in C/D

    // B fragments (W_hh.T), loop-invariant -> registers. For gate g the
    // N-tile base is g*64 + wave*16; per lane column n = g*64 + col.
    // B uses this code's contiguous k order, same as A: lanes 0-15 hold K=0..15, lanes 16-31 K=16..31.
    const _Float16* wp = s_W + col * kHH + 16 * kh;     // + g*64*kHH per gate
    v16h bfi0 = *(const v16h*)(wp + 0 * kHH * kHH + 0);
    v16h bfi1 = *(const v16h*)(wp + 0 * kHH * kHH + 32);
    v16h bff0 = *(const v16h*)(wp + 1 * kHH * kHH + 0);
    v16h bff1 = *(const v16h*)(wp + 1 * kHH * kHH + 32);
    v16h bfg0 = *(const v16h*)(wp + 2 * kHH * kHH + 0);
    v16h bfg1 = *(const v16h*)(wp + 2 * kHH * kHH + 32);
    v16h bfo0 = *(const v16h*)(wp + 3 * kHH * kHH + 0);
    v16h bfo1 = *(const v16h*)(wp + 3 * kHH * kHH + 32);

    // per-lane h-store constants (phys layout inverse map for k = col)
    const int wr_base = (col >> 5) * 512 + ((col >> 4) & 1) * 256
                      + (col & 15);

    const v8f zero8 = {};
    v8f c_reg;                            // cell state, fragment layout
#pragma unroll
    for (int j = 0; j < 8; ++j) c_reg[j] = 0.f;

    // one LSTM step: gates = table[tok] + h@Whh.T ; update c; h' -> s_wr
    auto step = [&](const int* ptok, const _Float16* s_rd, _Float16* s_wr) {
        // tokens for this step: 8 contiguous ints per lane (time-major layout)
        v4i tk0 = *(const v4i*)(ptok);
        v4i tk1 = *(const v4i*)(ptok + 4);

        // input-term gather (consumed in the elementwise phase, not as WMMA C)
        v4f qv[8];
#pragma unroll
        for (int j = 0; j < 8; ++j) {
            int tok = (j < 4) ? tk0[j & 3] : tk1[j & 3];
            qv[j] = *(const v4f*)(s_table + tok * kNG + col * 4);
        }

        // A fragment 0 (K 0..31): contiguous per-lane v16h in phys layout
        v16h a = *(const v16h*)(s_rd + lane * 16);
        v8f acc_i = __builtin_amdgcn_wmma_f32_16x16x32_f16(false, a, false, bfi0, (short)0, zero8, false, false);
        v8f acc_f = __builtin_amdgcn_wmma_f32_16x16x32_f16(false, a, false, bff0, (short)0, zero8, false, false);
        v8f acc_g = __builtin_amdgcn_wmma_f32_16x16x32_f16(false, a, false, bfg0, (short)0, zero8, false, false);
        v8f acc_o = __builtin_amdgcn_wmma_f32_16x16x32_f16(false, a, false, bfo0, (short)0, zero8, false, false);
        // A fragment 1 (K 32..63), reuse the same register tuple
        a = *(const v16h*)(s_rd + 512 + lane * 16);
        acc_i = __builtin_amdgcn_wmma_f32_16x16x32_f16(false, a, false, bfi1, (short)0, acc_i, false, false);
        acc_f = __builtin_amdgcn_wmma_f32_16x16x32_f16(false, a, false, bff1, (short)0, acc_f, false, false);
        acc_g = __builtin_amdgcn_wmma_f32_16x16x32_f16(false, a, false, bfg1, (short)0, acc_g, false, false);
        acc_o = __builtin_amdgcn_wmma_f32_16x16x32_f16(false, a, false, bfo1, (short)0, acc_o, false, false);

        // staged cell update: 8 independent chains per stage for ILP
#pragma unroll
        for (int j = 0; j < 8; ++j) acc_i[j] = sigmoidf_(acc_i[j] + qv[j][0]);
#pragma unroll
        for (int j = 0; j < 8; ++j) acc_f[j] = sigmoidf_(acc_f[j] + qv[j][1]);
#pragma unroll
        for (int j = 0; j < 8; ++j) acc_g[j] = tanhf_(acc_g[j] + qv[j][2]);
#pragma unroll
        for (int j = 0; j < 8; ++j) acc_o[j] = sigmoidf_(acc_o[j] + qv[j][3]);
#pragma unroll
        for (int j = 0; j < 8; ++j)
            c_reg[j] = acc_f[j] * c_reg[j] + acc_i[j] * acc_g[j];
#pragma unroll
        for (int j = 0; j < 8; ++j) {
            int m = j + 8 * kh;
            s_wr[wr_base + m * 16] = (_Float16)(acc_o[j] * tanhf_(c_reg[j]));
        }
        __syncthreads();
    };

    // direction folded into a strided token pointer: no per-step index selects
    const int  sdt  = dir ? -16 : 16;
    const int* ptok = s_tok + 8 * kh + (dir ? (kT - 1) * 16 : 0);
    for (int t = 0; t < kT; t += 2) {
        step(ptok,       s_h0, s_h1);   // even step: read h0, write h1
        step(ptok + sdt, s_h1, s_h0);   // odd  step: read h1, write h0
        ptok += 2 * sdt;
    }

    // kT is even -> final h lives in s_h0 (phys layout)
    for (int i = tid; i < 16 * kHH; i += 128) {
        int m = i >> 6, k = i & (kHH - 1);
        h_final[(dir * kB + b0 + m) * kHH + k] = s_h0[h_phys(m, k)];
    }
}

// ---------------------------------------------------------------------------
// out[b][v] = concat(h_f[b], h_b[b]) @ W_fc.T + b_fc   via WMMA, bias in C.
// grid B/16 x 256 threads (8 waves, one 16-col N-tile each, K=128 -> 4 WMMAs)
// s_hid staged in A-fragment physical layout -> contiguous v16h loads.
// ---------------------------------------------------------------------------
__global__ void blstm_fc(const _Float16* __restrict__ h_final,
                         const _Float16* __restrict__ Wfc,
                         const float* __restrict__ b_fc,
                         float* __restrict__ out) {
    __shared__ _Float16 s_hid[16 * 128];  // 4 frags x 512 halves, phys layout
    const int tid = threadIdx.x;
    const int b0  = blockIdx.x * 16;

    for (int i = tid; i < 16 * 128; i += 256) {
        int m = i >> 7, k = i & 127;
        _Float16 v = (k < kHH) ? h_final[(b0 + m) * kHH + k]
                               : h_final[(kB + b0 + m) * kHH + (k - kHH)];
        int frag = k >> 5, kk = k & 31;
        s_hid[frag * 512 + (m + 16 * ((kk >> 4) & 1)) * 16
              + (kk & 15)] = v;
    }
    __syncthreads();

    const int lane = tid & 31;
    const int wave = tid >> 5;
    const int ln16 = lane & 15;
    const int kh   = lane >> 4;
    const int n    = wave * 16 + ln16;

    v8f acc;
    const float bias = b_fc[n];
#pragma unroll
    for (int j = 0; j < 8; ++j) acc[j] = bias;

#pragma unroll
    for (int frag = 0; frag < 4; ++frag) {
        v16h a  = *(const v16h*)(s_hid + frag * 512 + lane * 16);
        v16h bf = *(const v16h*)(Wfc + n * 128 + frag * 32 + 16 * kh);
        acc = __builtin_amdgcn_wmma_f32_16x16x32_f16(false, a, false, bf, (short)0, acc, false, false);
    }

#pragma unroll
    for (int j = 0; j < 8; ++j) {
        int m = j + 8 * kh;
        out[(b0 + m) * kV + n] = acc[j];
    }
}

extern "C" void kernel_launch(void* const* d_in, const int* in_sizes, int n_in,
                              void* d_out, int out_size, void* d_ws, size_t ws_size,
                              hipStream_t stream) {
    (void)in_sizes; (void)n_in; (void)out_size; (void)ws_size;
    const int*   x     = (const int*)d_in[0];
    // d_in[1] = lengths : unused by the reference computation
    const float* emb   = (const float*)d_in[2];
    const float* Wih_f = (const float*)d_in[3];
    const float* Whh_f = (const float*)d_in[4];
    const float* Wih_b = (const float*)d_in[5];
    const float* Whh_b = (const float*)d_in[6];
    const float* Wfc   = (const float*)d_in[7];
    const float* bfc   = (const float*)d_in[8];
    float* out = (float*)d_out;

    char* ws = (char*)d_ws;
    float*    table_f = (float*)(ws + 0);        // 128*256 f32 = 131072 B
    float*    table_b = (float*)(ws + 131072);   // 131072 B
    _Float16* whh_f16 = (_Float16*)(ws + 262144);// 256*64 f16 = 32768 B
    _Float16* whh_b16 = (_Float16*)(ws + 294912);// 32768 B
    _Float16* wfc16   = (_Float16*)(ws + 327680);// 128*128 f16 = 32768 B
    _Float16* h_fin   = (_Float16*)(ws + 360448);// [2][1024][64] f16 = 262144 B

    blstm_prep_table<<<dim3(128, 2), 256, 0, stream>>>(emb, Wih_f, Wih_b, table_f, table_b);
    blstm_f32_to_f16<<<64, 256, 0, stream>>>(Whh_f, whh_f16, kNG * kHH);
    blstm_f32_to_f16<<<64, 256, 0, stream>>>(Whh_b, whh_b16, kNG * kHH);
    blstm_f32_to_f16<<<64, 256, 0, stream>>>(Wfc, wfc16, kV * kH);

    const size_t smem = 131072 + 32768 + 32768 + 2048 + 2048; // 200704 B
    blstm_scan<<<dim3(kB / 16, 2), 128, smem, stream>>>(x, table_f, table_b,
                                                        whh_f16, whh_b16, h_fin);
    blstm_fc<<<kB / 16, 256, 0, stream>>>(h_fin, wfc16, bfc, out);
}
